// SparseUNet3D_MDN_8589934592312
// MI455X (gfx1250) — matrix operand order repaired and checked by reading, not compiled, not run on hardware
//
#include <hip/hip_runtime.h>
#include <hip/hip_bf16.h>

typedef _Float16 h16;
typedef __attribute__((ext_vector_type(16))) _Float16 v16h;
typedef __attribute__((ext_vector_type(8)))  _Float16 v8h;
typedef __attribute__((ext_vector_type(8)))  float    v8f;

union AFrag { v16h v; v8h p[2]; };

// ---------------------------------------------------------------------------
// 3D conv as implicit GEMM with V_WMMA_F32_16X16X32_F16, register-blocked:
// one wave (wave32) computes a 32(voxel) x (16*NB)(cout) tile.
//   A layout (ISA 7.12.2, 16-bit A 16x32): lane%16 = row M; g = lane/16;
//     halves 0..7 -> K = g*8+0..7 ; halves 8..15 -> K = 16+g*8+0..7
//     -> two contiguous 16B loads per lane per 32-K chunk.
//   B layout (32x16): lane%16 = col N; halves h -> K=(h/8)*16+g*8+h%8 (same as A).
//     Pre-packed on device so each B fragment is one aligned 32B v16h load.
// Per K-chunk: 2 A gathers + NB B loads -> 2*NB WMMAs (16 FLOP/B at NB=2).
// Epilogue: fused BN + ReLU + sparsity-mask, store f16 activations.
// ---------------------------------------------------------------------------
template<int DO,int HO,int WO,int DI,int HI,int WI,int KS,int STR,int PD,
         int CIN,int CA,int COUT,int NB>
__global__ __launch_bounds__(32) void conv_wmma(
    const h16* __restrict__ inA, const h16* __restrict__ inB,
    const h16* __restrict__ wp,  const float* __restrict__ bn,
    const float* __restrict__ mask, h16* __restrict__ out)
{
  constexpr int TAPS = KS*KS*KS;
  constexpr int CH   = CIN/32;
  constexpr int CHA  = CA/32;             // chunks served by first buffer
  constexpr int CB   = CIN - CA;          // second (concat) buffer channels
  constexpr int BSTR = TAPS*CH*32;        // v16h elements per 16-cout tile
  const int lane = threadIdx.x;
  const int g  = lane >> 4;
  const int m  = lane & 15;
  const int mt = blockIdx.x;
  const int nt = blockIdx.y;

  const int vox0 = mt*32 + m;
  const int vox1 = vox0 + 16;
  const int d0 = vox0/(HO*WO), h0 = (vox0/WO)%HO, w0 = vox0%WO;
  const int d1 = vox1/(HO*WO), h1 = (vox1/WO)%HO, w1 = vox1%WO;

  v8f acc[2][NB] = {};
  const v16h* wbase = (const v16h*)wp + (size_t)(nt*NB)*BSTR + lane;
  const v16h az = {};

  int t = 0;
  for (int kd = 0; kd < KS; ++kd)
  for (int kh = 0; kh < KS; ++kh)
  for (int kw = 0; kw < KS; ++kw, ++t) {
    const int dn0 = d0*STR + kd - PD, hn0 = h0*STR + kh - PD, wn0 = w0*STR + kw - PD;
    const int dn1 = d1*STR + kd - PD, hn1 = h1*STR + kh - PD, wn1 = w1*STR + kw - PD;
    const bool inb0 = (unsigned)dn0 < (unsigned)DI && (unsigned)hn0 < (unsigned)HI &&
                      (unsigned)wn0 < (unsigned)WI;
    const bool inb1 = (unsigned)dn1 < (unsigned)DI && (unsigned)hn1 < (unsigned)HI &&
                      (unsigned)wn1 < (unsigned)WI;
    const size_t nv0 = ((size_t)dn0*HI + hn0)*WI + wn0;
    const size_t nv1 = ((size_t)dn1*HI + hn1)*WI + wn1;
    // hoisted per-tap bases; per-chunk channel offsets fold into load imm
    const size_t a0A = nv0*(size_t)CA + g*8, a0B = nv0*(size_t)CB + g*8;
    const size_t a1A = nv1*(size_t)CA + g*8, a1B = nv1*(size_t)CB + g*8;
    #pragma unroll
    for (int ck = 0; ck < CH; ++ck) {
      AFrag a0, a1; a0.v = az; a1.v = az;
      if (ck < CHA) {                      // compile-time under unroll
        const int co = ck*32;
        if (inb0) { a0.p[0] = *(const v8h*)(inA + a0A + co);
                    a0.p[1] = *(const v8h*)(inA + a0A + co + 16); }
        if (inb1) { a1.p[0] = *(const v8h*)(inA + a1A + co);
                    a1.p[1] = *(const v8h*)(inA + a1A + co + 16); }
      } else {
        const int co = (ck - CHA)*32;
        if (inb0) { a0.p[0] = *(const v8h*)(inB + a0B + co);
                    a0.p[1] = *(const v8h*)(inB + a0B + co + 16); }
        if (inb1) { a1.p[0] = *(const v8h*)(inB + a1B + co);
                    a1.p[1] = *(const v8h*)(inB + a1B + co + 16); }
      }
      #pragma unroll
      for (int b = 0; b < NB; ++b) {
        const v16h bm = wbase[(size_t)b*BSTR + (size_t)(t*CH + ck)*32];
        acc[0][b] = __builtin_amdgcn_wmma_f32_16x16x32_f16(false, a0.v, false, bm,
                                                           (short)0, acc[0][b], false, false);
        acc[1][b] = __builtin_amdgcn_wmma_f32_16x16x32_f16(false, a1.v, false, bm,
                                                           (short)0, acc[1][b], false, false);
      }
    }
  }
  // Epilogue: C/D layout -> VGPR r holds (M = r + 8*g, N = lane%16)
  #pragma unroll
  for (int b = 0; b < NB; ++b) {
    const int cout = (nt*NB + b)*16 + m;
    const float sc = bn[cout] * rsqrtf(bn[3*COUT + cout] + 1e-5f);
    const float mu = bn[2*COUT + cout];
    const float bt = bn[COUT + cout];
    #pragma unroll
    for (int mi = 0; mi < 2; ++mi) {
      #pragma unroll
      for (int r = 0; r < 8; ++r) {
        const int rv = mt*32 + mi*16 + g*8 + r;
        float y = (acc[mi][b][r] - mu)*sc + bt;
        y = fmaxf(y, 0.f) * mask[rv];
        out[(size_t)rv*COUT + cout] = (h16)y;
      }
    }
  }
}

// ---------------------------------------------------------------------------
// Transposed conv 2x2x2 stride 2 (VALID), tap-major (grid.z = 8): GEMM over
// the coarse grid with uniform per-wave B, scatter to strided fine voxels,
// multiply by fine mask. 32x32 tile per wave (MB=2, NB=2).
// ---------------------------------------------------------------------------
template<int DC,int HC,int WC,int CIN,int COUT>
__global__ __launch_bounds__(32) void upconv_wmma(
    const h16* __restrict__ in, const h16* __restrict__ wp,
    const float* __restrict__ maskF, h16* __restrict__ out)
{
  constexpr int CH = CIN/32;
  constexpr int HF = 2*HC, WF = 2*WC;
  constexpr int BSTR = 8*CH*32;                 // v16h per 16-cout tile
  const int lane = threadIdx.x;
  const int g  = lane >> 4;
  const int m  = lane & 15;
  const int mt = blockIdx.x;
  const int nt = blockIdx.y;
  const int t  = blockIdx.z;                    // tap 0..7
  const int kd = t >> 2, kh = (t >> 1) & 1, kw = t & 1;

  const int vox0 = mt*32 + m, vox1 = vox0 + 16;
  v8f acc[2][2] = {};
  const v16h* wbase = (const v16h*)wp + ((size_t)(nt*2)*8 + t)*CH*32 + lane;
  #pragma unroll
  for (int ck = 0; ck < CH; ++ck) {
    AFrag a0, a1;
    const size_t b0 = (size_t)vox0*CIN + ck*32 + g*8;
    const size_t b1 = (size_t)vox1*CIN + ck*32 + g*8;
    a0.p[0] = *(const v8h*)(in + b0); a0.p[1] = *(const v8h*)(in + b0 + 16);
    a1.p[0] = *(const v8h*)(in + b1); a1.p[1] = *(const v8h*)(in + b1 + 16);
    #pragma unroll
    for (int b = 0; b < 2; ++b) {
      const v16h bm = wbase[(size_t)b*BSTR + (size_t)ck*32];
      acc[0][b] = __builtin_amdgcn_wmma_f32_16x16x32_f16(false, a0.v, false, bm,
                                                         (short)0, acc[0][b], false, false);
      acc[1][b] = __builtin_amdgcn_wmma_f32_16x16x32_f16(false, a1.v, false, bm,
                                                         (short)0, acc[1][b], false, false);
    }
  }
  #pragma unroll
  for (int b = 0; b < 2; ++b) {
    const int cout = (nt*2 + b)*16 + m;
    #pragma unroll
    for (int mi = 0; mi < 2; ++mi) {
      #pragma unroll
      for (int r = 0; r < 8; ++r) {
        const int rv = mt*32 + mi*16 + g*8 + r;
        const int dc = rv / (HC*WC);
        const int hc = (rv / WC) % HC;
        const int wc = rv % WC;
        const size_t f = ((size_t)(2*dc + kd)*HF + (2*hc + kh))*WF + (2*wc + kw);
        out[f*COUT + cout] = (h16)(acc[mi][b][r] * maskF[f]);
      }
    }
  }
}

// ---------------------------------------------------------------------------
// MDN head as a WMMA GEMM: M = voxels, N = 45 padded to 48 (3 tiles), K = 32.
// Epilogue fuses +bias, *mask, clip, and the 5-way output scatter; all routing
// is branchless (cndmask selects), only the final store is predicated.
// ---------------------------------------------------------------------------
__global__ __launch_bounds__(32) void head_wmma(
    const h16* __restrict__ c0, const h16* __restrict__ wpk,
    const float* __restrict__ bh, const float* __restrict__ m0,
    float* __restrict__ out)
{
  constexpr size_t V = (size_t)32*128*128;
  const int lane = threadIdx.x, g = lane >> 4, m = lane & 15;
  const int mt = blockIdx.x;
  const int vox = mt*16 + m;
  AFrag a;
  const size_t base = (size_t)vox*32 + g*8;
  a.p[0] = *(const v8h*)(c0 + base);
  a.p[1] = *(const v8h*)(c0 + base + 16);
  const v16h* wv = (const v16h*)wpk + lane;
  v8f acc[3] = {};
  #pragma unroll
  for (int b = 0; b < 3; ++b) {
    const v16h bm = wv[(size_t)b*32];
    acc[b] = __builtin_amdgcn_wmma_f32_16x16x32_f16(false, a.v, false, bm,
                                                    (short)0, acc[b], false, false);
  }
  #pragma unroll
  for (int b = 0; b < 3; ++b) {
    const int j  = b*16 + m;              // 0..47; 45..47 are pad columns
    const int jj = (j < 45) ? j : 44;     // OOB-safe bias index (select)
    const float bias = bh[jj];
    const int k = jj / 9, t = jj % 9;
    const bool is_mu  = (t < 3);
    const bool is_sig = (t >= 3) && (t < 6);
    // offset pieces, all branchless:
    const size_t base15 = (is_mu ? (size_t)0 : 15*V) + (size_t)k*3 + (t % 3);
    const size_t base5  = (size_t)(30 + 5*(t - 6))*V + k;
    #pragma unroll
    for (int r = 0; r < 8; ++r) {
      const size_t rv = (size_t)mt*16 + g*8 + r;
      float s = (acc[b][r] + bias) * m0[rv];
      const float sc = fminf(fmaxf(s, -5.f), 3.f);
      s = is_sig ? sc : s;
      const size_t o = (t < 6) ? (base15 + rv*15) : (base5 + rv*5);
      if (j < 45) out[o] = s;             // single predicated store
    }
  }
}

// ---------------------------------------------------------------------------
// Pack raw (taps, CINraw, COUTraw) f32 weights into per-lane WMMA B-fragment
// register images: index (((nt*TAPS + t)*CH + ck)*32 + lane)*16 + h,
// cin = ck*32 + (h/8)*16 + (lane/16)*8 + h%8; zero-padded past CINraw / COUTraw.
// ---------------------------------------------------------------------------
__global__ void pack_wB(const float* __restrict__ wr, h16* __restrict__ wp,
                        int TAPS, int CINraw, int CINP, int COUTraw, int COUTP)
{
  const int CH = CINP/32;
  const size_t total = (size_t)(COUTP/16)*TAPS*CH*512;
  size_t i = (size_t)blockIdx.x*256 + threadIdx.x;
  if (i >= total) return;
  const int hh = (int)(i & 15); size_t r = i >> 4;
  const int lane = (int)(r & 31); r >>= 5;
  const int ck = (int)(r % CH); r /= CH;
  const int t  = (int)(r % TAPS);
  const int nt = (int)(r / TAPS);
  const int cin  = ck*32 + (hh >> 3)*16 + (lane >> 4)*8 + (hh & 7);
  const int cout = nt*16 + (lane & 15);
  const float v = (cin < CINraw && cout < COUTraw)
                ? wr[((size_t)t*CINraw + cin)*COUTraw + cout] : 0.f;
  wp[i] = (h16)v;
}

// Pack input x (f32, 20ch) into f16 activations padded to 32 channels.
__global__ void pack_x(const float* __restrict__ x, h16* __restrict__ a)
{
  size_t i = (size_t)blockIdx.x*256 + threadIdx.x;
  const int c = (int)(i & 31); const size_t v = i >> 5;
  a[i] = (c < 20) ? (h16)x[v*20 + c] : (h16)0.f;
}

// 2x2x2 max-pool of the occupancy mask.
template<int DC,int HC,int WC>
__global__ void maskpool(const float* __restrict__ mf, float* __restrict__ mc)
{
  const int i = blockIdx.x*256 + threadIdx.x;
  if (i >= DC*HC*WC) return;
  const int d = i/(HC*WC), h = (i/WC)%HC, w = i%WC;
  constexpr int HF = 2*HC, WF = 2*WC;
  float mx = 0.f;
  #pragma unroll
  for (int a = 0; a < 2; ++a)
  #pragma unroll
  for (int b = 0; b < 2; ++b)
  #pragma unroll
  for (int c = 0; c < 2; ++c)
    mx = fmaxf(mx, mf[((size_t)(2*d + a)*HF + (2*h + b))*WF + (2*w + c)]);
  mc[i] = mx;
}

// ---------------------------------------------------------------------------
// Workspace layout (compile-time; all offsets naturally 256B aligned).
// ---------------------------------------------------------------------------
constexpr size_t V0 = (size_t)32*128*128, V1 = (size_t)16*64*64,
                 V2 = (size_t)8*32*32,    V3 = (size_t)4*16*16;
constexpr size_t S0 = V0*32*2, S1 = V1*64*2, S2 = V2*128*2, S3 = V3*256*2;
constexpr size_t PW(size_t NT, size_t T, size_t CH){ return NT*T*CH*512*2; }

constexpr size_t O_ACTX = 0;
constexpr size_t O_U0   = O_ACTX;          // alias: actX dead after enc0
constexpr size_t O_E0   = O_ACTX + S0;
constexpr size_t O_C0   = O_E0 + S0;
constexpr size_t O_D0   = O_C0 + S0;
constexpr size_t O_E1   = O_D0 + S1;
constexpr size_t O_U1   = O_E1 + S1;
constexpr size_t O_C1   = O_U1 + S1;
constexpr size_t O_D1   = O_C1 + S1;
constexpr size_t O_E2   = O_D1 + S2;
constexpr size_t O_U2   = O_E2 + S2;
constexpr size_t O_C2   = O_U2 + S2;
constexpr size_t O_D2   = O_C2 + S2;
constexpr size_t O_BT   = O_D2 + S3;
constexpr size_t O_M1   = O_BT + S3;
constexpr size_t O_M2   = O_M1 + V1*4;
constexpr size_t O_M3   = O_M2 + V2*4;
constexpr size_t O_WENC0  = O_M3 + V3*4;
constexpr size_t O_WDOWN0 = O_WENC0  + PW(2,27,1);
constexpr size_t O_WENC1  = O_WDOWN0 + PW(4,8,1);
constexpr size_t O_WDOWN1 = O_WENC1  + PW(4,27,2);
constexpr size_t O_WENC2  = O_WDOWN1 + PW(8,8,2);
constexpr size_t O_WDOWN2 = O_WENC2  + PW(8,27,4);
constexpr size_t O_WBOTT  = O_WDOWN2 + PW(16,8,4);
constexpr size_t O_WUP2   = O_WBOTT  + PW(16,27,8);
constexpr size_t O_WDEC2  = O_WUP2   + PW(8,8,8);
constexpr size_t O_WUP1   = O_WDEC2  + PW(8,27,8);
constexpr size_t O_WDEC1  = O_WUP1   + PW(4,8,4);
constexpr size_t O_WUP0   = O_WDEC1  + PW(4,27,4);
constexpr size_t O_WDEC0  = O_WUP0   + PW(2,8,2);
constexpr size_t O_WHEAD  = O_WDEC0  + PW(2,27,2);

extern "C" void kernel_launch(void* const* d_in, const int* in_sizes, int n_in,
                              void* d_out, int out_size, void* d_ws, size_t ws_size,
                              hipStream_t stream) {
  (void)in_sizes; (void)n_in; (void)out_size; (void)ws_size;
  const float* x       = (const float*)d_in[0];
  const float* m0      = (const float*)d_in[1];
  const float* w_enc0  = (const float*)d_in[2];  const float* bn_enc0 = (const float*)d_in[3];
  const float* w_down0 = (const float*)d_in[4];  const float* bn_down0= (const float*)d_in[5];
  const float* w_enc1  = (const float*)d_in[6];  const float* bn_enc1 = (const float*)d_in[7];
  const float* w_down1 = (const float*)d_in[8];  const float* bn_down1= (const float*)d_in[9];
  const float* w_enc2  = (const float*)d_in[10]; const float* bn_enc2 = (const float*)d_in[11];
  const float* w_down2 = (const float*)d_in[12]; const float* bn_down2= (const float*)d_in[13];
  const float* w_bott  = (const float*)d_in[14]; const float* bn_bott = (const float*)d_in[15];
  const float* w_up2   = (const float*)d_in[16];
  const float* w_dec2  = (const float*)d_in[17]; const float* bn_dec2 = (const float*)d_in[18];
  const float* w_up1   = (const float*)d_in[19];
  const float* w_dec1  = (const float*)d_in[20]; const float* bn_dec1 = (const float*)d_in[21];
  const float* w_up0   = (const float*)d_in[22];
  const float* w_dec0  = (const float*)d_in[23]; const float* bn_dec0 = (const float*)d_in[24];
  const float* w_head  = (const float*)d_in[25]; const float* b_head  = (const float*)d_in[26];

  char* ws = (char*)d_ws;
  h16*  actX = (h16*)(ws + O_ACTX);
  h16  *e0=(h16*)(ws+O_E0), *u0=(h16*)(ws+O_U0), *c0=(h16*)(ws+O_C0);
  h16  *d0=(h16*)(ws+O_D0), *e1=(h16*)(ws+O_E1), *u1=(h16*)(ws+O_U1), *c1=(h16*)(ws+O_C1);
  h16  *d1=(h16*)(ws+O_D1), *e2=(h16*)(ws+O_E2), *u2=(h16*)(ws+O_U2), *c2=(h16*)(ws+O_C2);
  h16  *d2=(h16*)(ws+O_D2), *bt=(h16*)(ws+O_BT);
  float *m1=(float*)(ws+O_M1), *m2=(float*)(ws+O_M2), *m3=(float*)(ws+O_M3);

  auto packW = [&](const float* raw, size_t off, int TAPS, int CINraw, int CINP,
                   int COUTraw, int COUTP){
    size_t total = (size_t)(COUTP/16)*TAPS*(CINP/32)*512;
    pack_wB<<<dim3((unsigned)((total + 255)/256)), dim3(256), 0, stream>>>(
        raw, (h16*)(ws + off), TAPS, CINraw, CINP, COUTraw, COUTP);
  };
  packW(w_enc0 , O_WENC0 , 27, 20, 32, 32, 32);
  packW(w_down0, O_WDOWN0,  8, 32, 32, 64, 64);
  packW(w_enc1 , O_WENC1 , 27, 64, 64, 64, 64);
  packW(w_down1, O_WDOWN1,  8, 64, 64, 128, 128);
  packW(w_enc2 , O_WENC2 , 27, 128, 128, 128, 128);
  packW(w_down2, O_WDOWN2,  8, 128, 128, 256, 256);
  packW(w_bott , O_WBOTT , 27, 256, 256, 256, 256);
  packW(w_up2  , O_WUP2  ,  8, 256, 256, 128, 128);
  packW(w_dec2 , O_WDEC2 , 27, 256, 256, 128, 128);
  packW(w_up1  , O_WUP1  ,  8, 128, 128, 64, 64);
  packW(w_dec1 , O_WDEC1 , 27, 128, 128, 64, 64);
  packW(w_up0  , O_WUP0  ,  8, 64, 64, 32, 32);
  packW(w_dec0 , O_WDEC0 , 27, 64, 64, 32, 32);
  packW(w_head , O_WHEAD ,  1, 32, 32, 45, 48);

  maskpool<16,64,64><<<dim3((unsigned)((V1+255)/256)), 256, 0, stream>>>(m0, m1);
  maskpool<8,32,32> <<<dim3((unsigned)((V2+255)/256)), 256, 0, stream>>>(m1, m2);
  maskpool<4,16,16> <<<dim3((unsigned)((V3+255)/256)), 256, 0, stream>>>(m2, m3);

  pack_x<<<dim3((unsigned)(V0*32/256)), 256, 0, stream>>>(x, actX);

  const h16* wE0=(h16*)(ws+O_WENC0),  *wD0=(h16*)(ws+O_WDOWN0);
  const h16* wE1=(h16*)(ws+O_WENC1),  *wD1=(h16*)(ws+O_WDOWN1);
  const h16* wE2=(h16*)(ws+O_WENC2),  *wD2=(h16*)(ws+O_WDOWN2);
  const h16* wB =(h16*)(ws+O_WBOTT);
  const h16* wU2=(h16*)(ws+O_WUP2),   *wC2=(h16*)(ws+O_WDEC2);
  const h16* wU1=(h16*)(ws+O_WUP1),   *wC1=(h16*)(ws+O_WDEC1);
  const h16* wU0=(h16*)(ws+O_WUP0),   *wC0=(h16*)(ws+O_WDEC0);
  const h16* wHD=(h16*)(ws+O_WHEAD);

  // ---- encoder (32x32 tiles per wave; NB=2)
  conv_wmma<32,128,128,32,128,128,3,1,1, 32, 32, 32,2>
      <<<dim3((unsigned)(V0/32), 1), 32, 0, stream>>>(actX, actX, wE0, bn_enc0, m0, e0);
  conv_wmma<16, 64, 64,32,128,128,2,2,0, 32, 32, 64,2>
      <<<dim3((unsigned)(V1/32), 2), 32, 0, stream>>>(e0, e0, wD0, bn_down0, m1, d0);
  conv_wmma<16, 64, 64,16, 64, 64,3,1,1, 64, 64, 64,2>
      <<<dim3((unsigned)(V1/32), 2), 32, 0, stream>>>(d0, d0, wE1, bn_enc1, m1, e1);
  conv_wmma< 8, 32, 32,16, 64, 64,2,2,0, 64, 64,128,2>
      <<<dim3((unsigned)(V2/32), 4), 32, 0, stream>>>(e1, e1, wD1, bn_down1, m2, d1);
  conv_wmma< 8, 32, 32, 8, 32, 32,3,1,1,128,128,128,2>
      <<<dim3((unsigned)(V2/32), 4), 32, 0, stream>>>(d1, d1, wE2, bn_enc2, m2, e2);
  conv_wmma< 4, 16, 16, 8, 32, 32,2,2,0,128,128,256,2>
      <<<dim3((unsigned)(V3/32), 8), 32, 0, stream>>>(e2, e2, wD2, bn_down2, m3, d2);
  conv_wmma< 4, 16, 16, 4, 16, 16,3,1,1,256,256,256,2>
      <<<dim3((unsigned)(V3/32), 8), 32, 0, stream>>>(d2, d2, wB, bn_bott, m3, bt);

  // ---- decoder (up + concat-conv); concat via two-pointer channel split
  upconv_wmma<4,16,16,256,128>
      <<<dim3((unsigned)(V3/32), 4, 8), 32, 0, stream>>>(bt, wU2, m2, u2);
  conv_wmma< 8, 32, 32, 8, 32, 32,3,1,1,256,128,128,2>
      <<<dim3((unsigned)(V2/32), 4), 32, 0, stream>>>(u2, e2, wC2, bn_dec2, m2, c2);
  upconv_wmma<8,32,32,128,64>
      <<<dim3((unsigned)(V2/32), 2, 8), 32, 0, stream>>>(c2, wU1, m1, u1);
  conv_wmma<16, 64, 64,16, 64, 64,3,1,1,128, 64, 64,2>
      <<<dim3((unsigned)(V1/32), 2), 32, 0, stream>>>(u1, e1, wC1, bn_dec1, m1, c1);
  upconv_wmma<16,64,64,64,32>
      <<<dim3((unsigned)(V1/32), 1, 8), 32, 0, stream>>>(c1, wU0, m0, u0);
  conv_wmma<32,128,128,32,128,128,3,1,1, 64, 32, 32,2>
      <<<dim3((unsigned)(V0/32), 1), 32, 0, stream>>>(u0, e0, wC0, bn_dec0, m0, c0);

  // ---- MDN head as WMMA GEMM (N = 45 -> 48, 3 tiles)
  head_wmma<<<dim3((unsigned)(V0/16)), 32, 0, stream>>>(
      c0, wHD, b_head, m0, (float*)d_out);
}
